// Attention_30279519437373
// MI455X (gfx1250) — compile-verified, matrix operand order checked by reading, not run on hardware
//
#include <hip/hip_runtime.h>
#include <hip/hip_bf16.h>

typedef __attribute__((ext_vector_type(16))) _Float16 v16h;
typedef __attribute__((ext_vector_type(8)))  float    v8f;

#define BATCH 4
#define CHN   256
#define NPOS  4096          // 64*64
#define HEADS 4
#define DH    32
#define HID   128           // HEADS*DH
#define GROUPS 8
#define CPG    32           // CHN/GROUPS
#define EPSV   1e-5f
#define QSCALE 0.17677669529663687f   // 1/sqrt(32)

union AF { v16h v; _Float16 h[16]; uint4 q[2]; };

__device__ __forceinline__ v8f wmma_f16(v16h a, v16h b, v8f c) {
  return __builtin_amdgcn_wmma_f32_16x16x32_f16(false, a, false, b, (short)0, c, false, false);
}

// ---------------- K0: convert output-projection weights f32 -> f16 ----------------
__global__ void k_convert_wo(const float* __restrict__ wo, _Float16* __restrict__ woh) {
  int i = blockIdx.x * blockDim.x + threadIdx.x;
  if (i < CHN * HID) woh[i] = (_Float16)wo[i];
}

// ---------------- K1: GroupNorm statistics ----------------
// one block per (b, group); channels of a group are contiguous: 32*4096 floats
__global__ void k_gn_stats(const float* __restrict__ x, float* __restrict__ stats) {
  int bg = blockIdx.x;                         // 0..31
  const float* p = x + (size_t)bg * (CPG * NPOS);
  float s = 0.f, ss = 0.f;
  for (int i = threadIdx.x; i < CPG * NPOS; i += blockDim.x) {
    float v = p[i]; s += v; ss += v * v;
  }
  __shared__ float sh[512];
  sh[threadIdx.x] = s; sh[256 + threadIdx.x] = ss;
  __syncthreads();
  for (int st = 128; st > 0; st >>= 1) {
    if ((int)threadIdx.x < st) {
      sh[threadIdx.x] += sh[threadIdx.x + st];
      sh[256 + threadIdx.x] += sh[256 + threadIdx.x + st];
    }
    __syncthreads();
  }
  if (threadIdx.x == 0) {
    const float inv = 1.f / (float)(CPG * NPOS);
    float mean = sh[0] * inv;
    float var  = sh[256] * inv - mean * mean;
    stats[bg * 2 + 0] = mean;
    stats[bg * 2 + 1] = rsqrtf(var + EPSV);
  }
}

// ---------------- K2: fold GroupNorm affine into QKV weights ----------------
// weff[b,o,c] = wq[o,c]*rstd*gamma[c]  (f16);  qbias[b,o] = sum_c wq[o,c]*(beta[c]-mean*rstd*gamma[c])
__global__ void k_weff(const float* __restrict__ wq, const float* __restrict__ gamma,
                       const float* __restrict__ beta, const float* __restrict__ stats,
                       _Float16* __restrict__ weff, float* __restrict__ qbias) {
  int b = blockIdx.x / 384;
  int o = blockIdx.x % 384;
  int c = threadIdx.x;                         // 256 channels
  int g = c >> 5;
  float mean = stats[(b * GROUPS + g) * 2 + 0];
  float rstd = stats[(b * GROUPS + g) * 2 + 1];
  float w   = wq[o * CHN + c];
  float gsc = rstd * gamma[c];
  weff[((size_t)b * 384 + o) * CHN + c] = (_Float16)(w * gsc);
  __shared__ float sh[256];
  sh[c] = w * (beta[c] - mean * gsc);
  __syncthreads();
  for (int st = 128; st > 0; st >>= 1) {
    if (c < st) sh[c] += sh[c + st];
    __syncthreads();
  }
  if (c == 0) qbias[b * 384 + o] = sh[0];
}

// ---------------- K3: transpose + convert x -> xt[b][n][c] (f16), LDS-tiled ----------------
__global__ void k_transpose(const float* __restrict__ x, _Float16* __restrict__ xt) {
  __shared__ _Float16 sh[64][72];              // padded to dodge bank conflicts
  int blk = blockIdx.x;                        // 4 * 4 * 64 = 1024 tiles
  int b  = blk >> 8;
  int ct = (blk >> 6) & 3;
  int it = blk & 63;
  int c0 = ct * 64, i0 = it * 64;
  const float* xp = x + (size_t)b * CHN * NPOS;
#pragma unroll
  for (int rep = 0; rep < 16; rep++) {
    int idx = rep * 256 + threadIdx.x;
    int lc = idx >> 6, li = idx & 63;          // consecutive threads -> consecutive i (coalesced)
    sh[lc][li] = (_Float16)xp[(size_t)(c0 + lc) * NPOS + i0 + li];
  }
  __syncthreads();
  _Float16* op = xt + ((size_t)b * NPOS + i0) * CHN + c0;
#pragma unroll
  for (int rep = 0; rep < 16; rep++) {
    int idx = rep * 256 + threadIdx.x;
    int li = idx >> 6, lc = idx & 63;          // consecutive threads -> consecutive c (coalesced)
    op[(size_t)li * CHN + lc] = sh[lc][li];
  }
}

// ---------------- K4: QKV GEMM  C[384x4096] = weff[384x256] * xt^T + qbias ----------------
// all fragment loads are two contiguous b128s; scatter into q[b,h,n,32] (scaled), k[b,h,n,32], vT[b,h,32,n]
__global__ void k_qkv_gemm(const _Float16* __restrict__ weff, const float* __restrict__ qbias,
                           const _Float16* __restrict__ xt,
                           _Float16* __restrict__ qh, _Float16* __restrict__ kh,
                           _Float16* __restrict__ vth) {
  int wid  = (int)((blockIdx.x * blockDim.x + threadIdx.x) >> 5);   // 0..24575
  int lane = threadIdx.x & 31;
  int nt = wid & 255;
  int t  = wid >> 8;
  int mt = t % 24;
  int b  = t / 24;
  int m0 = mt * 16, n0 = nt * 16;
  int lrow = lane & 15;
  int lo   = (lane & 16) ? 8 : 0;
  int row = m0 + lrow;
  int col = n0 + lrow;
  const _Float16* A = weff + ((size_t)b * 384 + row) * CHN;
  const _Float16* Bm = xt + ((size_t)b * NPOS + col) * CHN;
  v8f acc = {};
  for (int k0 = 0; k0 < CHN; k0 += 32) {
    AF a, bf;
    a.q[0]  = *(const uint4*)(A + k0 + lo);
    a.q[1]  = *(const uint4*)(A + k0 + 16 + lo);
    bf.q[0] = *(const uint4*)(Bm + k0 + lo);
    bf.q[1] = *(const uint4*)(Bm + k0 + 16 + lo);
    acc = wmma_f16(a.v, bf.v, acc);
  }
  int mhalf = (lane >> 4) << 3;
#pragma unroll
  for (int r = 0; r < 8; r++) {
    int o = m0 + mhalf + r;                    // output channel 0..383
    float v = acc[r] + qbias[b * 384 + o];
    int head = (o >> 5) & 3;
    int d    = o & 31;
    size_t bh = (size_t)b * HEADS + head;
    if (o < 128)       qh[(bh * NPOS + col) * DH + d] = (_Float16)(v * QSCALE);
    else if (o < 256)  kh[(bh * NPOS + col) * DH + d] = (_Float16)v;
    else               vth[(bh * DH + d) * NPOS + col] = (_Float16)v;
  }
}

// ---------------- K5: flash attention ----------------
// 512 blocks (16 bh * 32 qtiles), 8 waves, 16 queries/wave; K/V staged in LDS 64 keys at a time
// via GLOBAL_LOAD_ASYNC_TO_LDS_B128 (ASYNCcnt), bypassing the register file.
__global__ void k_attention(const _Float16* __restrict__ qh, const _Float16* __restrict__ kh,
                            const _Float16* __restrict__ vth, _Float16* __restrict__ att) {
  __shared__ _Float16 kt[64 * 32];             // [key][d]   4KB
  __shared__ _Float16 vt[32 * 64];             // [d][j]     4KB
  __shared__ _Float16 pst[8 * 16 * 64];        // per-wave P 16KB
  int tid = threadIdx.x;
  int wv = tid >> 5, lane = tid & 31;
  int blk = blockIdx.x;
  int qt = blk & 31;
  int bh = blk >> 5;
  int q0 = qt * 128 + wv * 16;
  const _Float16* Q  = qh  + (size_t)bh * NPOS * DH;
  const _Float16* K  = kh  + (size_t)bh * NPOS * DH;
  const _Float16* VT = vth + (size_t)bh * DH * NPOS;
  _Float16* L = pst + wv * 16 * 64;
  int lrow = lane & 15;
  int lo   = (lane & 16) ? 8 : 0;
  int mrow = (lane >> 4) << 3;

  AF qa;
  { const _Float16* p = Q + (size_t)(q0 + lrow) * DH;
    qa.q[0] = *(const uint4*)(p + lo);
    qa.q[1] = *(const uint4*)(p + 16 + lo); }

  float mi[8], li[8];
  v8f o0 = {}, o1 = {};
#pragma unroll
  for (int r = 0; r < 8; r++) { mi[r] = -1e30f; li[r] = 0.f; }

  int kkey = tid >> 2, kseg = (tid & 3) << 3;  // cooperative K-tile indices
  int vd   = tid >> 3, vseg = (tid & 7) << 3;  // cooperative V-tile indices
  // LDS byte offsets for the async destinations (low 32 bits of generic shared addr)
  unsigned ldsk = (unsigned)(uintptr_t)(kt + kkey * 32 + kseg);
  unsigned ldsv = (unsigned)(uintptr_t)(vt + vd * 64 + vseg);

  for (int j0 = 0; j0 < NPOS; j0 += 64) {
    __syncthreads();                           // previous tile fully consumed
    {
      unsigned long long gk = (unsigned long long)(uintptr_t)(K + (size_t)(j0 + kkey) * DH + kseg);
      unsigned long long gv = (unsigned long long)(uintptr_t)(VT + (size_t)vd * NPOS + j0 + vseg);
      asm volatile("global_load_async_to_lds_b128 %0, %1, off"
                   :: "v"(ldsk), "v"(gk) : "memory");
      asm volatile("global_load_async_to_lds_b128 %0, %1, off"
                   :: "v"(ldsv), "v"(gv) : "memory");
      asm volatile("s_wait_asynccnt 0" ::: "memory");
    }
    __syncthreads();                           // tile visible to all waves

    v8f s[4];
#pragma unroll
    for (int jj = 0; jj < 4; jj++) {
      AF kb;
      const _Float16* p = kt + (jj * 16 + lrow) * 32;
      kb.q[0] = *(const uint4*)(p + lo);
      kb.q[1] = *(const uint4*)(p + 16 + lo);
      v8f z = {};
      s[jj] = wmma_f16(qa.v, kb.v, z);
    }

    // online softmax: C-fragment rows live in 16-lane groups (xor masks 1..8)
#pragma unroll
    for (int r = 0; r < 8; r++) {
      float x0 = s[0][r], x1 = s[1][r], x2 = s[2][r], x3 = s[3][r];
      float mx = fmaxf(fmaxf(x0, x1), fmaxf(x2, x3));
      for (int msk = 1; msk < 16; msk <<= 1) mx = fmaxf(mx, __shfl_xor(mx, msk, 32));
      float mn = fmaxf(mi[r], mx);
      float alpha = __expf(mi[r] - mn);
      float p0 = __expf(x0 - mn), p1 = __expf(x1 - mn);
      float p2 = __expf(x2 - mn), p3 = __expf(x3 - mn);
      float rs = (p0 + p1) + (p2 + p3);
      for (int msk = 1; msk < 16; msk <<= 1) rs += __shfl_xor(rs, msk, 32);
      li[r] = li[r] * alpha + rs;
      mi[r] = mn;
      o0[r] *= alpha;
      o1[r] *= alpha;
      int m = r + mrow;
      _Float16* lp = L + m * 64 + lrow;
      lp[0]  = (_Float16)p0;
      lp[16] = (_Float16)p1;
      lp[32] = (_Float16)p2;
      lp[48] = (_Float16)p3;
    }
    // P staging is wave-private; DS ops are in-order within a wave
    asm volatile("s_wait_dscnt 0" ::: "memory");
    AF pa0, pa1;
    { const _Float16* p = L + lrow * 64;
      pa0.q[0] = *(const uint4*)(p + lo);      pa0.q[1] = *(const uint4*)(p + 16 + lo);
      pa1.q[0] = *(const uint4*)(p + 32 + lo); pa1.q[1] = *(const uint4*)(p + 48 + lo); }
    { const _Float16* p = vt + lrow * 64;      // d = lrow
      AF va, vb;
      va.q[0] = *(const uint4*)(p + lo);       va.q[1] = *(const uint4*)(p + 16 + lo);
      vb.q[0] = *(const uint4*)(p + 32 + lo);  vb.q[1] = *(const uint4*)(p + 48 + lo);
      o0 = wmma_f16(pa0.v, va.v, o0);
      o0 = wmma_f16(pa1.v, vb.v, o0);
    }
    { const _Float16* p = vt + (16 + lrow) * 64;  // d = 16+lrow
      AF va, vb;
      va.q[0] = *(const uint4*)(p + lo);       va.q[1] = *(const uint4*)(p + 16 + lo);
      vb.q[0] = *(const uint4*)(p + 32 + lo);  vb.q[1] = *(const uint4*)(p + 48 + lo);
      o1 = wmma_f16(pa0.v, va.v, o1);
      o1 = wmma_f16(pa1.v, vb.v, o1);
    }
  }

  int b = bh >> 2, h = bh & 3;
#pragma unroll
  for (int r = 0; r < 8; r++) {
    int m = r + mrow;
    float inv = 1.f / li[r];
    size_t base = ((size_t)b * NPOS + q0 + m) * HID + h * DH;   // att[b][n][128]
    att[base + lrow]      = (_Float16)(o0[r] * inv);
    att[base + 16 + lrow] = (_Float16)(o1[r] * inv);
  }
}

// ---------------- K6: output projection  out[256x4096] = Wout[256x128] * att^T + b ----------------
__global__ void k_out_gemm(const _Float16* __restrict__ woh, const _Float16* __restrict__ att,
                           const float* __restrict__ bout, float* __restrict__ out) {
  int wid  = (int)((blockIdx.x * blockDim.x + threadIdx.x) >> 5);  // 0..16383
  int lane = threadIdx.x & 31;
  int nt = wid & 255;
  int mt = (wid >> 8) & 15;
  int b  = wid >> 12;
  int m0 = mt * 16, n0 = nt * 16;
  int lrow = lane & 15;
  int lo   = (lane & 16) ? 8 : 0;
  int row = m0 + lrow;
  int col = n0 + lrow;
  const _Float16* A  = woh + (size_t)row * HID;
  const _Float16* Bm = att + ((size_t)b * NPOS + col) * HID;   // [n][128]: contiguous over K
  v8f acc = {};
#pragma unroll
  for (int k0 = 0; k0 < HID; k0 += 32) {
    AF a, bf;
    a.q[0]  = *(const uint4*)(A + k0 + lo);
    a.q[1]  = *(const uint4*)(A + k0 + 16 + lo);
    bf.q[0] = *(const uint4*)(Bm + k0 + lo);
    bf.q[1] = *(const uint4*)(Bm + k0 + 16 + lo);
    acc = wmma_f16(a.v, bf.v, acc);
  }
#pragma unroll
  for (int r = 0; r < 8; r++) {
    int o = m0 + r + ((lane >> 4) << 3);
    out[((size_t)b * CHN + o) * NPOS + col] = acc[r] + bout[o];
  }
}

// ---------------- launch ----------------
extern "C" void kernel_launch(void* const* d_in, const int* in_sizes, int n_in,
                              void* d_out, int out_size, void* d_ws, size_t ws_size,
                              hipStream_t stream) {
  const float* x     = (const float*)d_in[0];   // [4,256,64,64]
  const float* wqkv  = (const float*)d_in[1];   // [384,256]
  const float* wout  = (const float*)d_in[2];   // [256,128]
  const float* bout  = (const float*)d_in[3];   // [256]
  const float* gamma = (const float*)d_in[4];   // [256]
  const float* beta  = (const float*)d_in[5];   // [256]
  float* out = (float*)d_out;

  _Float16* ws   = (_Float16*)d_ws;
  _Float16* woh  = ws;                       //  32768 halfs
  _Float16* weff = ws + 32768;               //  4*384*256  = 393216
  _Float16* xt   = ws + 425984;              //  4*4096*256 = 4194304
  _Float16* qh   = ws + 4620288;             //  4*4*4096*32 = 2097152
  _Float16* kh   = ws + 6717440;             //  2097152
  _Float16* vth  = ws + 8814592;             //  2097152
  _Float16* att  = ws + 10911744;            //  4*4096*128 = 2097152
  float* qbias   = (float*)(ws + 13008896);  //  1536 floats
  float* stats   = qbias + 1536;             //  64 floats

  k_convert_wo<<<128, 256, 0, stream>>>(wout, woh);
  k_gn_stats<<<32, 256, 0, stream>>>(x, stats);
  k_weff<<<1536, 256, 0, stream>>>(wqkv, gamma, beta, stats, weff, qbias);
  k_transpose<<<1024, 256, 0, stream>>>(x, xt);
  k_qkv_gemm<<<3072, 256, 0, stream>>>(weff, qbias, xt, qh, kh, vth);
  k_attention<<<512, 256, 0, stream>>>(qh, kh, vth, att);
  k_out_gemm<<<2048, 256, 0, stream>>>(woh, att, bout, out);
}
